// ScaledDotProductAttention_46007689675101
// MI455X (gfx1250) — matrix operand order repaired and checked by reading, not compiled, not run on hardware
//
#include <hip/hip_runtime.h>

// B=16, D=64, S=2048 attention, softmax over the QUERY axis (dim=1).
//   K1: q,k (f32 [b][d][s]) -> bf16 [b][s][d]  (async global->LDS + transpose)
//   K2: v   (f32 [b][d][s]) -> bf16 [b][d][s]
//   K3: rsum[b][t] = 1 / sum_s exp(mask ? -1000 : qk/8)    (WMMA bf16)
//   K4: out[b][s][d] = sum_t exp(...)*rsum[t] * v[d][t]    (WMMA bf16, t-split waves)

#define NB 16
#define ND 64
#define NS 2048

// 0.125 * log2(e), and -1000 * log2(e) for masked entries (exp(x) == exp2(x*log2e))
#define SC_L2E 0.18033688011112042f
#define MASK_L2E -1442.6950408889634f

typedef __attribute__((ext_vector_type(16))) __bf16 bf16x16;
typedef __attribute__((ext_vector_type(8)))  __bf16 bf16x8;
typedef __attribute__((ext_vector_type(8)))  float  f32x8;

static __device__ __forceinline__ bf16x16 load_a_frag(const __bf16* __restrict__ row,
                                                      int kbase, int half) {
  // A-matrix 16x32 bf16: lane (m = l&15, half = l>>4):
  //   e=0..15 -> K = kbase + half*16 + e (32B contiguous)
  union { bf16x16 v; bf16x8 h[2]; } u;
  u.h[0] = *(const bf16x8*)(row + kbase + half * 16);
  u.h[1] = *(const bf16x8*)(row + kbase + half * 16 + 8);
  return u.v;
}

static __device__ __forceinline__ f32x8 wmma_bf16(bf16x16 a, bf16x16 b, f32x8 c) {
  return __builtin_amdgcn_wmma_f32_16x16x32_bf16(false, a, false, b, (short)0, c,
                                                 false, false);
}

static __device__ __forceinline__ unsigned mbyte(uint2 mm, int r) {
  return (((r < 4) ? mm.x : mm.y) >> (8 * (r & 3))) & 0xFFu;
}

// ---------------- K1: transpose + convert q/k -> bf16 [b][s][d] ----------------
// Read phase uses the gfx1250 async global->LDS path (per-lane LDS destinations
// let us keep the padded f32 tile); write phase converts + stores coalesced.
__global__ void tcvt_qk_kernel(const float* __restrict__ q, const float* __restrict__ k,
                               __bf16* __restrict__ qT, __bf16* __restrict__ kT) {
  __shared__ float tf[ND][68];  // pad 4 floats -> row stride 272B (16B aligned)
  const int b = blockIdx.y;
  const int s0 = blockIdx.x * 64;
  const float* src = (blockIdx.z == 0) ? q : k;
  __bf16* dst = (blockIdx.z == 0) ? qT : kT;
  const int t = threadIdx.x;  // 256 threads
  const float* gbase = src + ((size_t)(b * ND)) * NS + s0;
#pragma unroll
  for (int i = 0; i < 4; ++i) {
    int c = i * 256 + t;            // 1024 chunks of 16B
    int d = c >> 4, sb = c & 15;    // row d, 4-float chunk sb
    const float* gp = gbase + (size_t)d * NS + sb * 4;
    unsigned lp = (unsigned)(uintptr_t)(&tf[d][sb * 4]);
    asm volatile("global_load_async_to_lds_b128 %0, %1, off"
                 :: "v"(lp), "v"(gp) : "memory");
  }
  asm volatile("s_wait_asynccnt 0x0" ::: "memory");
  __syncthreads();
#pragma unroll
  for (int i = 0; i < 16; ++i) {
    int idx = i * 256 + t;
    int s = idx >> 6, d = idx & 63;  // write coalesced along d
    dst[((size_t)(b * NS + s0 + s)) * ND + d] = (__bf16)tf[d][s];
  }
}

// ---------------- K2: convert v -> bf16 [b][d][s] ----------------
__global__ void cvt_v_kernel(const float* __restrict__ v, __bf16* __restrict__ vb) {
  int i = blockIdx.x * 256 + threadIdx.x;
  vb[i] = (__bf16)v[i];
}

// ---------------- K3: reciprocal column sums of exp(scores) ----------------
// One wave per 16 t-columns; loops all s-tiles. scores^T tile (t x s) =
// A(kT 16x32) x B(qT 32x16), K = d (two chunks of 32).
__global__ void colsum_kernel(const __bf16* __restrict__ qT, const __bf16* __restrict__ kT,
                              const unsigned char* __restrict__ mask,
                              float* __restrict__ rsum) {
  const int b = blockIdx.y;
  const int wave = threadIdx.x >> 5;
  const int lane = threadIdx.x & 31;
  const int t0 = (blockIdx.x * 8 + wave) * 16;
  const int m = lane & 15, half = lane >> 4;

  const __bf16* kTb = kT + (size_t)b * NS * ND;
  const __bf16* qTb = qT + (size_t)b * NS * ND;
  const unsigned char* mb = mask + (size_t)b * NS * NS;

  const __bf16* krow = kTb + (size_t)(t0 + m) * ND;
  const bf16x16 a0 = load_a_frag(krow, 0, half);   // fixed over the s loop
  const bf16x16 a1 = load_a_frag(krow, 32, half);

  f32x8 acc = {};
  for (int s0 = 0; s0 < NS; s0 += 16) {
    int s0p = (s0 + 16 < NS) ? s0 + 16 : 0;  // next-iteration prefetch (wrapped)
    __builtin_prefetch(qTb + (size_t)(s0p + m) * ND, 0, 3);
    __builtin_prefetch(mb + (size_t)(s0p + m) * NS + t0, 0, 3);

    const __bf16* qrow = qTb + (size_t)(s0 + m) * ND;  // B-frag: n = lane&15
    bf16x16 b0 = *(const bf16x16*)(qrow + half * 16);
    bf16x16 b1 = *(const bf16x16*)(qrow + 32 + half * 16);
    f32x8 c = {};
    c = wmma_bf16(a0, b0, c);
    c = wmma_bf16(a1, b1, c);
    // lane holds col s = s0+m, rows t = t0 + 8*half + r
    uint2 mm = *(const uint2*)(mb + (size_t)(s0 + m) * NS + t0 + 8 * half);
#pragma unroll
    for (int r = 0; r < 8; ++r) {
      float a = c[r] * SC_L2E;
      if (mbyte(mm, r)) a = MASK_L2E;
      acc[r] += __builtin_amdgcn_exp2f(a);
    }
  }
  // reduce across the 16 lanes of each half (xor widths stay inside the half)
#pragma unroll
  for (int off = 1; off < 16; off <<= 1) {
#pragma unroll
    for (int r = 0; r < 8; ++r) acc[r] += __shfl_xor(acc[r], off, 32);
  }
  if (m == 0) {
#pragma unroll
    for (int r = 0; r < 8; ++r)
      rsum[(size_t)b * NS + t0 + 8 * half + r] = 1.0f / acc[r];
  }
}

// ---------------- K4: out = softmax-normalized attn @ V^T ----------------
// Wave pair splits the t-range (no redundant GEMM work, 2x parallelism);
// partials combined through LDS. Each wave: 2 s-tiles x full D (4 d-tiles).
__global__ void attn_out_kernel(const __bf16* __restrict__ qT, const __bf16* __restrict__ kT,
                                const __bf16* __restrict__ vb,
                                const unsigned char* __restrict__ mask,
                                const float* __restrict__ rsum, float* __restrict__ out) {
  __shared__ float lds_acc[4][32][64];  // [local s-group][lane][ns*32+dt*8+r] = 32KB
  const int b = blockIdx.y;
  const int wave = threadIdx.x >> 5;
  const int lane = threadIdx.x & 31;
  const int sgl = wave >> 1;  // local s-group 0..3
  const int th = wave & 1;    // t-half 0/1
  const int s0 = (blockIdx.x * 4 + sgl) * 32;  // 2 s-tiles per wave
  const int m = lane & 15, half = lane >> 4;
  const bool lohalf = (half == 0);
  const int tbeg = th * (NS / 2), tend = tbeg + NS / 2;

  const __bf16* qTb = qT + (size_t)b * NS * ND;
  const __bf16* kTb = kT + (size_t)b * NS * ND;
  const __bf16* vbb = vb + (size_t)b * ND * NS;
  const unsigned char* mb = mask + (size_t)b * NS * NS;
  const float* rs = rsum + (size_t)b * NS;

  bf16x16 qb[2][2];
#pragma unroll
  for (int ns = 0; ns < 2; ++ns) {
    const __bf16* qrow = qTb + (size_t)(s0 + ns * 16 + m) * ND;
    qb[ns][0] = *(const bf16x16*)(qrow + half * 16);
    qb[ns][1] = *(const bf16x16*)(qrow + 32 + half * 16);
  }

  f32x8 acc[2][4] = {};  // [s-tile][d-tile], out^T tiles (d x s)

  for (int t0 = tbeg; t0 < tend; t0 += 32) {
    int t0p = (t0 + 32 < tend) ? t0 + 32 : tbeg;  // next-iteration prefetch (wrapped)
    __builtin_prefetch(kTb + (size_t)(t0p + m) * ND, 0, 3);
    __builtin_prefetch(kTb + (size_t)(t0p + 16 + m) * ND, 0, 3);
#pragma unroll
    for (int dt = 0; dt < 4; ++dt)
      __builtin_prefetch(vbb + (size_t)(dt * 16 + m) * NS + t0p, 0, 3);
    __builtin_prefetch(mb + (size_t)(s0 + m) * NS + t0p, 0, 3);
    __builtin_prefetch(mb + (size_t)(s0 + 16 + m) * NS + t0p, 0, 3);

    const __bf16* krow1 = kTb + (size_t)(t0 + m) * ND;
    const __bf16* krow2 = kTb + (size_t)(t0 + 16 + m) * ND;
    bf16x16 ka[2][2];
    ka[0][0] = load_a_frag(krow1, 0, half);
    ka[0][1] = load_a_frag(krow1, 32, half);
    ka[1][0] = load_a_frag(krow2, 0, half);
    ka[1][1] = load_a_frag(krow2, 32, half);

    bf16x16 va[4];
#pragma unroll
    for (int dt = 0; dt < 4; ++dt) {
      const __bf16* vrow = vbb + (size_t)(dt * 16 + m) * NS;
      va[dt] = load_a_frag(vrow, t0, half);  // K = t chunk of 32
    }

#pragma unroll
    for (int ns = 0; ns < 2; ++ns) {
      f32x8 c1 = {}, c2 = {};
      c1 = wmma_bf16(ka[0][0], qb[ns][0], c1);
      c1 = wmma_bf16(ka[0][1], qb[ns][1], c1);
      c2 = wmma_bf16(ka[1][0], qb[ns][0], c2);
      c2 = wmma_bf16(ka[1][1], qb[ns][1], c2);

      const unsigned char* mrow = mb + (size_t)(s0 + ns * 16 + m) * NS + t0;
      uint2 mm1 = *(const uint2*)(mrow + 8 * half);
      uint2 mm2 = *(const uint2*)(mrow + 16 + 8 * half);

      float p1[8], p2[8];
#pragma unroll
      for (int r = 0; r < 8; ++r) {
        float a1 = c1[r] * SC_L2E;
        if (mbyte(mm1, r)) a1 = MASK_L2E;
        p1[r] = __builtin_amdgcn_exp2f(a1) * rs[t0 + 8 * half + r];
        float a2 = c2[r] * SC_L2E;
        if (mbyte(mm2, r)) a2 = MASK_L2E;
        p2[r] = __builtin_amdgcn_exp2f(a2) * rs[t0 + 16 + 8 * half + r];
      }
      // Build attn^T B-frag (32t x 16s): upper/lower half exchange.
      float sw1[8], sw2[8];
#pragma unroll
      for (int r = 0; r < 8; ++r) {
        sw1[r] = __shfl_xor(p1[r], 16, 32);
        sw2[r] = __shfl_xor(p2[r], 16, 32);
      }
      bf16x16 pb;
#pragma unroll
      for (int vv = 0; vv < 4; ++vv) {
        pb[2 * vv]     = (__bf16)(lohalf ? p1[2 * vv]     : sw2[2 * vv]);
        pb[2 * vv + 1] = (__bf16)(lohalf ? p1[2 * vv + 1] : sw2[2 * vv + 1]);
      }
#pragma unroll
      for (int vv = 4; vv < 8; ++vv) {
        int r = 2 * (vv - 4);
        pb[2 * vv]     = (__bf16)(lohalf ? sw1[r]     : p2[r]);
        pb[2 * vv + 1] = (__bf16)(lohalf ? sw1[r + 1] : p2[r + 1]);
      }
#pragma unroll
      for (int dt = 0; dt < 4; ++dt) acc[ns][dt] = wmma_bf16(va[dt], pb, acc[ns][dt]);
    }
  }

  // Combine the two t-halves through LDS (deterministic order), then store.
  if (th == 1) {
#pragma unroll
    for (int ns = 0; ns < 2; ++ns)
#pragma unroll
      for (int dt = 0; dt < 4; ++dt)
#pragma unroll
        for (int r = 0; r < 8; ++r)
          lds_acc[sgl][lane][ns * 32 + dt * 8 + r] = acc[ns][dt][r];
  }
  __syncthreads();
  if (th == 0) {
#pragma unroll
    for (int ns = 0; ns < 2; ++ns) {
#pragma unroll
      for (int dt = 0; dt < 4; ++dt)
#pragma unroll
        for (int r = 0; r < 8; ++r)
          acc[ns][dt][r] += lds_acc[sgl][lane][ns * 32 + dt * 8 + r];
    }
    // acc lane holds col s = s0+ns*16+(lane&15), rows d = dt*16+8*half+r
#pragma unroll
    for (int ns = 0; ns < 2; ++ns) {
      float* orow = out + (size_t)(b * NS + s0 + ns * 16 + m) * ND;
#pragma unroll
      for (int dt = 0; dt < 4; ++dt) {
        float4 lo = make_float4(acc[ns][dt][0], acc[ns][dt][1], acc[ns][dt][2], acc[ns][dt][3]);
        float4 hi = make_float4(acc[ns][dt][4], acc[ns][dt][5], acc[ns][dt][6], acc[ns][dt][7]);
        *(float4*)(orow + dt * 16 + 8 * half) = lo;
        *(float4*)(orow + dt * 16 + 8 * half + 4) = hi;
      }
    }
  }
}

extern "C" void kernel_launch(void* const* d_in, const int* in_sizes, int n_in,
                              void* d_out, int out_size, void* d_ws, size_t ws_size,
                              hipStream_t stream) {
  const float* q = (const float*)d_in[0];
  const float* k = (const float*)d_in[1];
  const float* v = (const float*)d_in[2];
  const unsigned char* mask = (const unsigned char*)d_in[3];
  float* out = (float*)d_out;

  const size_t n_el = (size_t)NB * NS * ND;  // 2,097,152
  __bf16* qT = (__bf16*)d_ws;
  __bf16* kT = qT + n_el;
  __bf16* vb = kT + n_el;
  float* rsum = (float*)(vb + n_el);  // NB*NS floats

  tcvt_qk_kernel<<<dim3(NS / 64, NB, 2), 256, 0, stream>>>(q, k, qT, kT);
  cvt_v_kernel<<<dim3((unsigned)(n_el / 256)), 256, 0, stream>>>(v, vb);
  colsum_kernel<<<dim3(NS / (16 * 8), NB), 256, 0, stream>>>(qT, kT, mask, rsum);
  attn_out_kernel<<<dim3(NS / (32 * 4), NB), 256, 0, stream>>>(qT, kT, vb, mask, rsum, out);
}
